// RNN_custom_9423158247575
// MI455X (gfx1250) — matrix operand order repaired and checked by reading, not compiled, not run on hardware
//
#include <hip/hip_runtime.h>

// ---------------------------------------------------------------------------
// CDNA5 (gfx1250) spiking-RNN forward.
//   - wave32, WMMA f32_16x16x32_f16
//   - 16 workgroups x 512 threads (16 waves); each WG owns 32 batch rows and
//     runs the whole T=100 recurrence with ONE workgroup barrier per step
//     (ping-pong spike buffers in LDS).
//   - pre-pass kernels: f32->f16 weight conversion, x [B,S,T] -> [T,B,S] f16.
// ---------------------------------------------------------------------------

typedef __attribute__((ext_vector_type(16))) _Float16     v16h;
typedef __attribute__((ext_vector_type(8)))  float        v8f;
typedef __attribute__((ext_vector_type(4)))  unsigned int v4u;

#define RDIM  512
#define BDIM  512
#define TDIM  100
#define ODIM  10
#define LDSTR 520   // f16 row stride (padded; 1040B rows stay 16B-aligned)

union FragU { v16h h; v4u u[2]; };

// B fragment: 32x16 f16, B[k][n] = W[n][k]; W row-major [512][512] f16.
// lane n = n_row; VGPR0-3: K = k0 + half*8 .. +7 ; VGPR4-7: K = k0 + 16 + half*8 .. +7.
__device__ __forceinline__ v16h load_B16(const _Float16* __restrict__ W,
                                         int n, int k0, int half) {
  FragU f;
  const _Float16* p = W + ((size_t)n << 9) + k0 + half * 8;
  f.u[0] = *(const v4u*)(p);
  f.u[1] = *(const v4u*)(p + 16);
  return f.h;
}

// A fragment: 16x32 f16 from row-major [.,512] global (row = batch row).
// VGPR0-3: K = k0 + half*8 .. +7 ; VGPR4-7: K = k0 + 16 + half*8 .. +7.
__device__ __forceinline__ v16h load_A_g(const _Float16* __restrict__ base,
                                         int row, int k0, int half) {
  FragU f;
  const _Float16* p = base + ((size_t)row << 9) + k0 + half * 8;
  f.u[0] = *(const v4u*)(p);
  f.u[1] = *(const v4u*)(p + 16);
  return f.h;
}

// Same, but from LDS with padded stride.
__device__ __forceinline__ v16h load_A_lds(const _Float16* p0,
                                           int row, int k0, int half) {
  FragU f;
  const _Float16* p = p0 + row * LDSTR + k0 + half * 8;
  f.u[0] = *(const v4u*)(p);
  f.u[1] = *(const v4u*)(p + 16);
  return f.h;
}

#define WMMA(a, b, c) \
  __builtin_amdgcn_wmma_f32_16x16x32_f16(false, (a), false, (b), (short)0, (c), false, false)

// -------------------------- pre-pass kernels -------------------------------

__global__ __launch_bounds__(256) void cvt_f32_f16(const float* __restrict__ in,
                                                   _Float16* __restrict__ out, int n) {
  int i = blockIdx.x * 256 + threadIdx.x;
  if (i < n) out[i] = (_Float16)in[i];
}

// x [B=512][S=512][T=100] f32  ->  xT [T][B*S] f16 (coalesced on both sides via LDS)
__global__ __launch_bounds__(256) void transpose_x(const float* __restrict__ x,
                                                   _Float16* __restrict__ xT) {
  __shared__ _Float16 tile[TDIM * 257];
  const int tid = threadIdx.x;
  const size_t idx = (size_t)blockIdx.x * 256 + tid;     // flattened b*512+s
  const float* src = x + idx * TDIM;                     // T contiguous
  for (int t = 0; t < TDIM; ++t) tile[t * 257 + tid] = (_Float16)src[t];
  __syncthreads();
  for (int i = tid; i < TDIM * 256; i += 256) {
    int t = i >> 8, sl = i & 255;
    xT[(size_t)t * (BDIM * RDIM) + (size_t)blockIdx.x * 256 + sl] = tile[t * 257 + sl];
  }
}

// ----------------------------- main kernel ---------------------------------

__global__ __launch_bounds__(512, 1)
void snn_main(const _Float16* __restrict__ xT,
              const float* __restrict__ r1_init, const float* __restrict__ r2_init,
              const _Float16* __restrict__ i2h16, const _Float16* __restrict__ h2h16,
              const _Float16* __restrict__ h2d16,
              const float* __restrict__ i2h_b, const float* __restrict__ h2h_b,
              const float* __restrict__ h2d_b,
              const float* __restrict__ d2o_w, const float* __restrict__ d2o_b,
              const float* __restrict__ tau_adp_r1, const float* __restrict__ tau_adp_r2,
              const float* __restrict__ tau_m_r1, const float* __restrict__ tau_m_r2,
              float* __restrict__ out)
{
  // ping-pong r1-spike buffers for this WG's 32 batch rows (66.5 KB total)
  __shared__ _Float16 lds_sp[2][32 * LDSTR];

  const int tid     = threadIdx.x;
  const int wid     = tid >> 5;     // 0..15 waves
  const int lane    = tid & 31;
  const int lane_lo = lane & 15;
  const int half    = lane >> 4;
  const int m0      = blockIdx.x * 32;

  // per-lane, per-ntile neuron parameters (n = column owned in C layout)
  float alpha1v[2], ro1v[2], bias1v[2], alpha2v[2], ro2v[2], bias2v[2];
  // neuron state in C-fragment layout: [ntile][mtile][vgpr]
  float mem1[2][2][8], b1[2][2][8], mem2[2][2][8], b2[2][2][8], sp2[2][2][8];

#pragma unroll
  for (int nt = 0; nt < 2; ++nt) {
    const int n = ((wid * 2 + nt) << 4) + lane_lo;
    alpha1v[nt] = __expf(-1.0f / tau_m_r1[n]);
    ro1v[nt]    = __expf(-1.0f / tau_adp_r1[n]);
    bias1v[nt]  = i2h_b[n] + h2h_b[n];
    alpha2v[nt] = __expf(-1.0f / tau_m_r2[n]);
    ro2v[nt]    = __expf(-1.0f / tau_adp_r2[n]);
    bias2v[nt]  = h2d_b[n];
#pragma unroll
    for (int mt = 0; mt < 2; ++mt)
#pragma unroll
      for (int r = 0; r < 8; ++r) {
        const int row = m0 + mt * 16 + r + half * 8;   // C layout M index
        mem1[nt][mt][r] = r1_init[(size_t)row * RDIM + n];
        b1[nt][mt][r]   = 0.01f;
        mem2[nt][mt][r] = r2_init[(size_t)row * RDIM + n];
        b2[nt][mt][r]   = 0.01f;
        sp2[nt][mt][r]  = r2_init[(size_t)row * RDIM + n];  // carry0: spike = init
      }
  }
  // initial r1 "spikes" = r1_init into buffer 0
  for (int i = tid; i < 32 * RDIM; i += 512) {
    const int row = i >> 9, col = i & (RDIM - 1);
    lds_sp[0][row * LDSTR + col] = (_Float16)r1_init[(size_t)(m0 + row) * RDIM + col];
  }
  __syncthreads();

  for (int t = 0; t < TDIM; ++t) {
    const _Float16* xt = xT + (size_t)t * (BDIM * RDIM);
    const _Float16* Rb = lds_sp[t & 1];         // old spikes (read)
    _Float16*       Wb = lds_sp[(t + 1) & 1];   // new spikes (write)

    // ---- layer 1: h_in = x_t@i2h.T + sp1@h2h.T + b ; neuron update ;
    //      new spikes go straight to the write buffer (race-free: all
    //      possible concurrent accesses to Rb are reads; Wb was last read
    //      before the previous step's barrier). ----
#pragma unroll
    for (int nt = 0; nt < 2; ++nt) {
      const int n0 = (wid * 2 + nt) << 4;
      const int n  = n0 + lane_lo;
      v8f acc0 = {}, acc1 = {};
#pragma unroll 2
      for (int k0 = 0; k0 < RDIM; k0 += 32) {
        v16h b  = load_B16(i2h16, n, k0, half);
        v16h a0 = load_A_g(xt, m0 + lane_lo,      k0, half);
        v16h a1 = load_A_g(xt, m0 + 16 + lane_lo, k0, half);
        acc0 = WMMA(a0, b, acc0);
        acc1 = WMMA(a1, b, acc1);
      }
#pragma unroll 2
      for (int k0 = 0; k0 < RDIM; k0 += 32) {
        v16h b  = load_B16(h2h16, n, k0, half);
        v16h a0 = load_A_lds(Rb, lane_lo,      k0, half);
        v16h a1 = load_A_lds(Rb, 16 + lane_lo, k0, half);
        acc0 = WMMA(a0, b, acc0);
        acc1 = WMMA(a1, b, acc1);
      }
#pragma unroll
      for (int mt = 0; mt < 2; ++mt) {
        v8f acc = mt ? acc1 : acc0;
#pragma unroll
        for (int r = 0; r < 8; ++r) {
          const int el    = (mt * 16 + r + half * 8) * LDSTR + n;
          const float hin = acc[r] + bias1v[nt];
          const float spo = (float)Rb[el];
          const float bb  = ro1v[nt] * b1[nt][mt][r] + (1.0f - ro1v[nt]) * spo;
          const float Bth = 0.3f + 1.8f * bb;
          const float mm  = mem1[nt][mt][r] * alpha1v[nt] + hin - Bth * spo;
          b1[nt][mt][r]   = bb;
          mem1[nt][mt][r] = mm;
          Wb[el] = (_Float16)((mm - Bth) > 0.0f ? 1.0f : 0.0f);
        }
      }
    }
    __syncthreads();   // single barrier per step: new spikes visible

    // ---- layer 2: d_in = sp1_new@h2d.T + b ; r2 state lane-private ----
#pragma unroll
    for (int nt = 0; nt < 2; ++nt) {
      const int n = ((wid * 2 + nt) << 4) + lane_lo;
      v8f acc0 = {}, acc1 = {};
#pragma unroll 2
      for (int k0 = 0; k0 < RDIM; k0 += 32) {
        v16h b  = load_B16(h2d16, n, k0, half);
        v16h a0 = load_A_lds(Wb, lane_lo,      k0, half);
        v16h a1 = load_A_lds(Wb, 16 + lane_lo, k0, half);
        acc0 = WMMA(a0, b, acc0);
        acc1 = WMMA(a1, b, acc1);
      }
#pragma unroll
      for (int mt = 0; mt < 2; ++mt) {
        v8f acc = mt ? acc1 : acc0;
#pragma unroll
        for (int r = 0; r < 8; ++r) {
          const float din = acc[r] + bias2v[nt];
          const float spo = sp2[nt][mt][r];
          const float bb  = ro2v[nt] * b2[nt][mt][r] + (1.0f - ro2v[nt]) * spo;
          const float Bth = 0.3f + 1.8f * bb;
          const float mm  = mem2[nt][mt][r] * alpha2v[nt] + din - Bth * spo;
          b2[nt][mt][r]   = bb;
          mem2[nt][mt][r] = mm;
          sp2[nt][mt][r]  = (mm - Bth) > 0.0f ? 1.0f : 0.0f;
        }
      }
    }
    // no trailing barrier: the next step's barrier protects the buffer swap
  }

  // ---- final projection: out = sp2 @ d2o.T + d2o_b (last timestep only) ----
  __syncthreads();               // last r2-GEMM reads done; reuse buffer 0
#pragma unroll
  for (int nt = 0; nt < 2; ++nt) {
    const int n = ((wid * 2 + nt) << 4) + lane_lo;
#pragma unroll
    for (int mt = 0; mt < 2; ++mt)
#pragma unroll
      for (int r = 0; r < 8; ++r)
        lds_sp[0][(mt * 16 + r + half * 8) * LDSTR + n] = (_Float16)sp2[nt][mt][r];
  }
  __syncthreads();
  if (tid < 32 * ODIM) {
    const int m = tid / ODIM, o = tid % ODIM;
    float s = d2o_b[o];
    for (int k = 0; k < RDIM; ++k)
      s += (float)lds_sp[0][m * LDSTR + k] * d2o_w[o * RDIM + k];
    out[(size_t)(m0 + m) * ODIM + o] = s;
  }
}

// ------------------------------ launcher -----------------------------------

extern "C" void kernel_launch(void* const* d_in, const int* in_sizes, int n_in,
                              void* d_out, int out_size, void* d_ws, size_t ws_size,
                              hipStream_t stream) {
  const float* x          = (const float*)d_in[0];
  const float* r1_init    = (const float*)d_in[1];
  const float* r2_init    = (const float*)d_in[2];
  const float* i2h_w      = (const float*)d_in[3];
  const float* i2h_b      = (const float*)d_in[4];
  const float* h2h_w      = (const float*)d_in[5];
  const float* h2h_b      = (const float*)d_in[6];
  const float* h2d_w      = (const float*)d_in[7];
  const float* h2d_b      = (const float*)d_in[8];
  const float* d2o_w      = (const float*)d_in[9];
  const float* d2o_b      = (const float*)d_in[10];
  const float* tau_adp_r1 = (const float*)d_in[11];
  const float* tau_adp_r2 = (const float*)d_in[12];
  const float* tau_m_r1   = (const float*)d_in[13];
  const float* tau_m_r2   = (const float*)d_in[14];

  const int W = RDIM * RDIM;                 // 262144
  _Float16* i2h16 = (_Float16*)d_ws;
  _Float16* h2h16 = i2h16 + W;
  _Float16* h2d16 = h2h16 + W;
  _Float16* xT    = h2d16 + W;               // [T][B*S] f16, ~52 MB

  cvt_f32_f16<<<(W + 255) / 256, 256, 0, stream>>>(i2h_w, i2h16, W);
  cvt_f32_f16<<<(W + 255) / 256, 256, 0, stream>>>(h2h_w, h2h16, W);
  cvt_f32_f16<<<(W + 255) / 256, 256, 0, stream>>>(h2d_w, h2d16, W);
  transpose_x<<<(BDIM * RDIM) / 256, 256, 0, stream>>>(x, xT);

  snn_main<<<BDIM / 32, 512, 0, stream>>>(xT, r1_init, r2_init,
                                          i2h16, h2h16, h2d16,
                                          i2h_b, h2h_b, h2d_b,
                                          d2o_w, d2o_b,
                                          tau_adp_r1, tau_adp_r2,
                                          tau_m_r1, tau_m_r2,
                                          (float*)d_out);
}
